// GatedMoE_83631603188334
// MI455X (gfx1250) — matrix operand order repaired and checked by reading, not compiled, not run on hardware
//
#include <hip/hip_runtime.h>

// GatedMoE (noisy top-2, SwiGLU experts) for gfx1250 / MI455X.
// Routed (top-2 only) f16-WMMA GEMMs with f32 accumulate.
//
// Sizes fixed by the reference:
#define BT   8192      // tokens
#define DD   1024      // model dim
#define HH   4096      // intermediate dim
#define EE   8         // experts
#define KKE  2         // top-k
#define PCAP 16896     // BT*KKE + EE*64 padding, 64-aligned expert regions

typedef __attribute__((ext_vector_type(16))) _Float16 v16h;
typedef __attribute__((ext_vector_type(8)))  _Float16 h8;
typedef __attribute__((ext_vector_type(8)))  float    v8f;
typedef __attribute__((ext_vector_type(4)))  float    f4;
typedef int i4v __attribute__((vector_size(16)));   // matches builtin's V4i

union Frag { v16h v; h8 h[2]; };

// gfx1250 async global->LDS copy (CDNA5 ASYNCcnt path), guarded so the build
// cannot break if the toolchain lacks the builtin.
#if defined(__HIP_DEVICE_COMPILE__) && __has_builtin(__builtin_amdgcn_global_load_async_to_lds_b128)
#define USE_ASYNC_LDS 1
#else
#define USE_ASYNC_LDS 0
#endif

__device__ __forceinline__ void wait_asynccnt0() {
#if defined(__HIP_DEVICE_COMPILE__)
#if __has_builtin(__builtin_amdgcn_s_wait_asynccnt)
  __builtin_amdgcn_s_wait_asynccnt(0);
#else
  asm volatile("s_wait_asynccnt 0" ::: "memory");
#endif
#endif
}

// load 8 f32 (two b128) and convert to 8 f16 (one b128 worth)
__device__ __forceinline__ h8 cvt8(const float* __restrict__ src) {
  f4 a = *(const f4*)(src);
  f4 b = *(const f4*)(src + 4);
  h8 r;
  r[0] = (_Float16)a.x; r[1] = (_Float16)a.y; r[2] = (_Float16)a.z; r[3] = (_Float16)a.w;
  r[4] = (_Float16)b.x; r[5] = (_Float16)b.y; r[6] = (_Float16)b.z; r[7] = (_Float16)b.w;
  return r;
}

// ---------------- 1) gating logits ----------------
__global__ void gate_logits_kernel(const float* __restrict__ x,
                                   const float* __restrict__ noise,
                                   const float* __restrict__ wg,
                                   const float* __restrict__ wn,
                                   float* __restrict__ logits) {
  int t = blockIdx.x * blockDim.x + threadIdx.x;   // t = b*EE + e
  if (t >= BT * EE) return;
  int b = t >> 3, e = t & 7;
  const f4* xr = (const f4*)(x + (size_t)b * DD);
  const f4* gr = (const f4*)(wg + (size_t)e * DD);
  const f4* nr = (const f4*)(wn + (size_t)e * DD);
  float sg = 0.f, sn = 0.f;
  for (int i = 0; i < DD / 4; ++i) {
    f4 a = xr[i], g = gr[i], n = nr[i];
    sg += a.x * g.x + a.y * g.y + a.z * g.z + a.w * g.w;
    sn += a.x * n.x + a.y * n.y + a.z * n.z + a.w * n.w;
  }
  float sp = (sn > 20.f) ? sn : log1pf(__expf(sn));   // softplus
  logits[t] = sg + noise[t] * sp;
}

// ---------------- 2) top-2 + softmax + counts ----------------
__global__ void topk_kernel(const float* __restrict__ logits,
                            float* __restrict__ gates_topk,
                            int* __restrict__ expert_topk,
                            int* __restrict__ counts) {
  int b = blockIdx.x * blockDim.x + threadIdx.x;
  if (b >= BT) return;
  float v[EE];
#pragma unroll
  for (int i = 0; i < EE; ++i) v[i] = logits[b * EE + i];
  int i0 = 0;
#pragma unroll
  for (int i = 1; i < EE; ++i) if (v[i] > v[i0]) i0 = i;   // first-index ties
  int i1 = (i0 == 0) ? 1 : 0;
#pragma unroll
  for (int i = 0; i < EE; ++i) if (i != i0 && v[i] > v[i1]) i1 = i;
  float ex = __expf(v[i1] - v[i0]);
  float p0 = 1.f / (1.f + ex);
  gates_topk[b * 2 + 0] = p0;
  gates_topk[b * 2 + 1] = ex * p0;
  expert_topk[b * 2 + 0] = i0;
  expert_topk[b * 2 + 1] = i1;
  atomicAdd(&counts[i0], 1);
  atomicAdd(&counts[i1], 1);
}

// ---------------- 3) exclusive scan (64-aligned regions) ----------------
__global__ void scan_kernel(const int* __restrict__ counts, int* __restrict__ offsets) {
  if (threadIdx.x == 0 && blockIdx.x == 0) {
    int o = 0;
    for (int e = 0; e < EE; ++e) { offsets[e] = o; o += (counts[e] + 63) & ~63; }
    offsets[EE] = o;
  }
}

// ---------------- 4) fill compacted slot lists ----------------
__global__ void fill_kernel(const float* __restrict__ gates_topk,
                            const int* __restrict__ expert_topk,
                            const int* __restrict__ offsets,
                            int* __restrict__ cursor,
                            int* __restrict__ slot_token,
                            float* __restrict__ slot_gate,
                            int* __restrict__ slot_of_token) {
  int b = blockIdx.x * blockDim.x + threadIdx.x;
  if (b >= BT) return;
#pragma unroll
  for (int k = 0; k < KKE; ++k) {
    int e = expert_topk[b * 2 + k];
    int pos = atomicAdd(&cursor[e], 1);
    int s = offsets[e] + pos;
    slot_token[s] = b;
    slot_gate[s] = gates_topk[b * 2 + k];
    slot_of_token[b * 2 + k] = s;
  }
}

// ---------------- 5) hbuf[slot,H] = silu(x@Wg^T) * (x@Wu^T)  (f16 out) ----
// block = 256 thr = 8 waves; block tile M=64 slots x N=64 of H; K-step 32 of D
__global__ __launch_bounds__(256) void moe_h_kernel(
    const float* __restrict__ x, const float* __restrict__ w_gate,
    const float* __restrict__ w_up, const int* __restrict__ offsets,
    const int* __restrict__ slot_token, _Float16* __restrict__ hbuf) {
  __shared__ _Float16 lds_a[64][32];
  __shared__ _Float16 lds_bg[64][32];
  __shared__ _Float16 lds_bu[64][32];

  const int slot_base = blockIdx.x * 64;
  const int nbase = blockIdx.y * 64;
  int e = EE - 1;
#pragma unroll
  for (int i = 0; i < EE; ++i)
    if (slot_base >= offsets[i] && slot_base < offsets[i + 1]) { e = i; break; }
  const float* wgm = w_gate + (size_t)e * HH * DD;
  const float* wum = w_up   + (size_t)e * HH * DD;

  const int tid = threadIdx.x;
  const int lrow = tid >> 2;          // 0..63
  const int lseg = (tid & 3) * 8;     // 0,8,16,24 (element offset)
  const int tok = slot_token[slot_base + lrow];
  const bool valid = (tok >= 0);
  const float* xrow = x + (size_t)(valid ? tok : 0) * DD;
  const float* wgrow = wgm + (size_t)(nbase + lrow) * DD + lseg;
  const float* wurow = wum + (size_t)(nbase + lrow) * DD + lseg;

  const int lane = tid & 31;
  const int wv = tid >> 5;            // 0..7
  const int mt = wv >> 1;             // M-tile 0..3
  const int nh = (wv & 1) * 32;       // N half 0 / 32
  const int lm = lane & 15;
  const int hi = lane >> 4;           // lane group

  v8f accg[2] = {}, accu[2] = {};

  for (int kb = 0; kb < DD; kb += 32) {
    h8 av = {};
    if (valid) av = cvt8(xrow + kb + lseg);
    *(h8*)&lds_a[lrow][lseg]  = av;
    *(h8*)&lds_bg[lrow][lseg] = cvt8(wgrow + kb);
    *(h8*)&lds_bu[lrow][lseg] = cvt8(wurow + kb);
    if (kb + 32 < DD) {
      __builtin_prefetch(wgrow + kb + 32, 0, 1);
      __builtin_prefetch(wurow + kb + 32, 0, 1);
    }
    __syncthreads();

    // A fragment: lane<16 holds K{0..15}, lane>=16 holds K{16..31}
    Frag af;
    const int arow = mt * 16 + lm;
    const int ab = hi * 16;
    af.h[0] = *(const h8*)&lds_a[arow][ab];
    af.h[1] = *(const h8*)&lds_a[arow][ab + 8];

#pragma unroll
    for (int nt = 0; nt < 2; ++nt) {
      const int brow = nh + nt * 16 + lm;
      const int bb = hi * 16;                       // K group 0..15 / 16..31
      Frag bg;
      bg.h[0] = *(const h8*)&lds_bg[brow][bb];
      bg.h[1] = *(const h8*)&lds_bg[brow][bb + 8];
      accg[nt] = __builtin_amdgcn_wmma_f32_16x16x32_f16(
          false, af.v, false, bg.v, (short)0, accg[nt], false, false);
      Frag bu;
      bu.h[0] = *(const h8*)&lds_bu[brow][bb];
      bu.h[1] = *(const h8*)&lds_bu[brow][bb + 8];
      accu[nt] = __builtin_amdgcn_wmma_f32_16x16x32_f16(
          false, af.v, false, bu.v, (short)0, accu[nt], false, false);
    }
    __syncthreads();
  }

  // epilogue: h = silu(g)*u ; C layout: elem r -> M = r + 8*hi, N = lm
  const int mrow0 = slot_base + mt * 16 + hi * 8;
#pragma unroll
  for (int nt = 0; nt < 2; ++nt) {
    const int col = nbase + nh + nt * 16 + lm;
#pragma unroll
    for (int r = 0; r < 8; ++r) {
      float g = accg[nt][r];
      float u = accu[nt][r];
      float hval = (g / (1.f + __expf(-g))) * u;
      hbuf[(size_t)(mrow0 + r) * HH + col] = (_Float16)hval;
    }
  }
}

// ---------------- 6) outpair[slot,D] = gate * (hbuf @ Wd^T) ----------------
__global__ __launch_bounds__(256) void moe_down_kernel(
    const _Float16* __restrict__ hbuf, const float* __restrict__ w_down,
    const int* __restrict__ offsets, const float* __restrict__ slot_gate,
    float* __restrict__ outpair) {
  __shared__ _Float16 lds_a[64][32];
  __shared__ _Float16 lds_b[64][32];
  __shared__ float lds_gate[64];

  const int slot_base = blockIdx.x * 64;
  const int dbase = blockIdx.y * 64;
  int e = EE - 1;
#pragma unroll
  for (int i = 0; i < EE; ++i)
    if (slot_base >= offsets[i] && slot_base < offsets[i + 1]) { e = i; break; }
  const float* wd = w_down + (size_t)e * DD * HH;

  const int tid = threadIdx.x;
  const int lrow = tid >> 2;
  const int lseg = (tid & 3) * 8;
  if (tid < 64) lds_gate[tid] = slot_gate[slot_base + tid];

  const _Float16* arow_g = hbuf + (size_t)(slot_base + lrow) * HH + lseg;
  const float*    brow_g = wd + (size_t)(dbase + lrow) * HH + lseg;

  const int lane = tid & 31;
  const int wv = tid >> 5;
  const int mt = wv >> 1;
  const int nh = (wv & 1) * 32;
  const int lm = lane & 15;
  const int hi = lane >> 4;

  v8f acc[2] = {};

  for (int kb = 0; kb < HH; kb += 32) {
    // A tile (hbuf) is already f16: use CDNA5 async global->LDS DMA when available.
#if USE_ASYNC_LDS
    __builtin_amdgcn_global_load_async_to_lds_b128(
        (__attribute__((address_space(1))) i4v*)(arow_g + kb),
        (__attribute__((address_space(3))) i4v*)&lds_a[lrow][lseg],
        0, 0);
#else
    *(h8*)&lds_a[lrow][lseg] = *(const h8*)(arow_g + kb);   // already f16
#endif
    *(h8*)&lds_b[lrow][lseg] = cvt8(brow_g + kb);
    if (kb + 32 < HH) __builtin_prefetch(brow_g + kb + 32, 0, 1);
#if USE_ASYNC_LDS
    wait_asynccnt0();
#endif
    __syncthreads();

    Frag af;
    const int arow = mt * 16 + lm;
    const int ab = hi * 16;
    af.h[0] = *(const h8*)&lds_a[arow][ab];
    af.h[1] = *(const h8*)&lds_a[arow][ab + 8];

#pragma unroll
    for (int nt = 0; nt < 2; ++nt) {
      const int brow = nh + nt * 16 + lm;
      const int bb = hi * 16;
      Frag bf;
      bf.h[0] = *(const h8*)&lds_b[brow][bb];
      bf.h[1] = *(const h8*)&lds_b[brow][bb + 8];
      acc[nt] = __builtin_amdgcn_wmma_f32_16x16x32_f16(
          false, af.v, false, bf.v, (short)0, acc[nt], false, false);
    }
    __syncthreads();
  }

  const int mloc = mt * 16 + hi * 8;
#pragma unroll
  for (int nt = 0; nt < 2; ++nt) {
    const int col = dbase + nh + nt * 16 + lm;
#pragma unroll
    for (int r = 0; r < 8; ++r) {
      float v = acc[nt][r] * lds_gate[mloc + r];
      outpair[(size_t)(slot_base + mloc + r) * DD + col] = v;
    }
  }
}

// ---------------- 7) combine (fixed order => deterministic) ----------------
__global__ void combine_kernel(const float* __restrict__ outpair,
                               const int* __restrict__ slot_of_token,
                               float* __restrict__ out) {
  int i = blockIdx.x * blockDim.x + threadIdx.x;   // over BT*DD/4
  if (i >= BT * DD / 4) return;
  int b = i / (DD / 4);
  int c = (i % (DD / 4)) * 4;
  int s0 = slot_of_token[b * 2 + 0];
  int s1 = slot_of_token[b * 2 + 1];
  f4 a = *(const f4*)(outpair + (size_t)s0 * DD + c);
  f4 d = *(const f4*)(outpair + (size_t)s1 * DD + c);
  *(f4*)(out + (size_t)b * DD + c) = a + d;
}

extern "C" void kernel_launch(void* const* d_in, const int* in_sizes, int n_in,
                              void* d_out, int out_size, void* d_ws, size_t ws_size,
                              hipStream_t stream) {
  const float* x      = (const float*)d_in[0];
  const float* noise  = (const float*)d_in[1];
  const float* wg     = (const float*)d_in[2];
  const float* wn     = (const float*)d_in[3];
  const float* w_gate = (const float*)d_in[4];
  const float* w_up   = (const float*)d_in[5];
  const float* w_down = (const float*)d_in[6];
  float* out = (float*)d_out;

  char* p = (char*)d_ws;
  size_t used = 0;
  auto alloc = [&](size_t bytes) -> char* {
    char* r = p + used;
    used += (bytes + 255) & ~(size_t)255;
    return r;
  };
  float* logits       = (float*)alloc((size_t)BT * EE * 4);
  float* gates_topk   = (float*)alloc((size_t)BT * KKE * 4);
  int*   expert_topk  = (int*)  alloc((size_t)BT * KKE * 4);
  int*   counts       = (int*)  alloc(EE * 4);
  int*   offsets      = (int*)  alloc((EE + 1) * 4);
  int*   cursor       = (int*)  alloc(EE * 4);
  int*   slot_token   = (int*)  alloc((size_t)PCAP * 4);
  float* slot_gate    = (float*)alloc((size_t)PCAP * 4);
  int*   slot_of_tok  = (int*)  alloc((size_t)BT * KKE * 4);
  _Float16* hbuf      = (_Float16*)alloc((size_t)PCAP * HH * 2);
  float* outpair      = (float*)alloc((size_t)PCAP * DD * 4);
  if (used > ws_size) return;  // workspace too small; bail (avoid OOB)

  (void)hipMemsetAsync(counts, 0, EE * 4, stream);
  (void)hipMemsetAsync(cursor, 0, EE * 4, stream);
  (void)hipMemsetAsync(slot_token, 0xFF, (size_t)PCAP * 4, stream);   // -1 => pad
  (void)hipMemsetAsync(slot_gate, 0, (size_t)PCAP * 4, stream);

  gate_logits_kernel<<<(BT * EE) / 256, 256, 0, stream>>>(x, noise, wg, wn, logits);
  topk_kernel<<<BT / 256, 256, 0, stream>>>(logits, gates_topk, expert_topk, counts);
  scan_kernel<<<1, 1, 0, stream>>>(counts, offsets);
  fill_kernel<<<BT / 256, 256, 0, stream>>>(gates_topk, expert_topk, offsets, cursor,
                                            slot_token, slot_gate, slot_of_tok);

  dim3 g2(PCAP / 64, HH / 64);   // 264 x 64
  moe_h_kernel<<<g2, 256, 0, stream>>>(x, w_gate, w_up, offsets, slot_token, hbuf);

  dim3 g3(PCAP / 64, DD / 64);   // 264 x 16
  moe_down_kernel<<<g3, 256, 0, stream>>>(hbuf, w_down, offsets, slot_gate, outpair);

  combine_kernel<<<(BT * DD / 4) / 256, 256, 0, stream>>>(outpair, slot_of_tok, out);
}
